// Money_former_DINT_62234076119139
// MI455X (gfx1250) — matrix operand order repaired and checked by reading, compiled, not run on hardware
//
#include <hip/hip_runtime.h>
#include <hip/hip_bf16.h>
#include <math.h>

// ---------------- model constants ----------------
#define D_MODEL   1024
#define NHEAD     16
#define NUM_LAYERS 6
#define D_FF      2816
#define IN_FEAT   64
#define N_PRED    4
#define HD        64
#define HD2       32
#define BATCH     2
#define T_TOK     512            // 1 sep + 511 inputs
#define SCALING_C 0.17677669529663687f   // 1/sqrt(32)
#define EPS_DEF   1.1920928955078125e-07f
#define EPS_HEAD  1e-05f

typedef __attribute__((ext_vector_type(16))) __bf16 v16bf;
typedef __attribute__((ext_vector_type(8)))  float  v8f;

// ---------------- block reductions ----------------
__device__ __forceinline__ float blk_sum(float v, float* red, int tid) {
    red[tid] = v; __syncthreads();
    for (int s = 128; s > 0; s >>= 1) {
        if (tid < s) red[tid] += red[tid + s];
        __syncthreads();
    }
    float r = red[0]; __syncthreads();
    return r;
}
__device__ __forceinline__ float blk_max(float v, float* red, int tid) {
    red[tid] = v; __syncthreads();
    for (int s = 128; s > 0; s >>= 1) {
        if (tid < s) red[tid] = fmaxf(red[tid], red[tid + s]);
        __syncthreads();
    }
    float r = red[0]; __syncthreads();
    return r;
}

// convert 4 consecutive fragment elements from one float4
#define CVT4(dst, base, v4)                 \
    do {                                    \
        dst[(base) + 0] = (__bf16)(v4).x;   \
        dst[(base) + 1] = (__bf16)(v4).y;   \
        dst[(base) + 2] = (__bf16)(v4).z;   \
        dst[(base) + 3] = (__bf16)(v4).w;   \
    } while (0)

// ---------------- WMMA GEMM: C[M,N] = A[M,K] @ W[N,K]^T + bias (+res) ----------------
// One wave32 per 16x64 strip (4 N-tiles share one A fragment).
// Ping-pong double-buffered K loop (K must be a multiple of 64; all call sites
// use K = 1024 or 2816): buf1 loads issue while buf0 feeds the 4 WMMAs, and
// vice versa, with two fixed register sets so no v_mov shuffling is needed.
struct Stage {
    float4 ra[4];
    float4 rb0[4], rb1[4], rb2[4], rb3[4];
};

template<bool HAS_RES>
__device__ __forceinline__ void store_tile(const v8f& acc, float* __restrict__ C,
                                           const float* __restrict__ res, float bv,
                                           int mbase, int n, int N)
{
    #pragma unroll
    for (int v = 0; v < 8; ++v) {
        size_t idx = (size_t)(mbase + v) * N + n;
        float val = acc[v] + bv;
        if (HAS_RES) val += res[idx];
        C[idx] = val;
    }
}

template<bool HAS_RES>
__global__ __launch_bounds__(128)
void gemm_wmma_bf16(const float* __restrict__ A, const float* __restrict__ W,
                    const float* __restrict__ bias, const float* __restrict__ res,
                    float* __restrict__ C, int M, int N, int K)
{
    int wave    = blockIdx.x * (blockDim.x >> 5) + (threadIdx.x >> 5);
    int stripsN = N >> 6;                      // strips of 64 columns
    int strips  = (M >> 4) * stripsN;
    if (wave >= strips) return;
    int tm = wave / stripsN;
    int ts = wave % stripsN;

    int lane = threadIdx.x & 31;
    int g    = lane >> 4;          // half-wave group
    int r    = lane & 15;          // A: row within tile / B: col within tile
    int g2   = g << 1;             // unused: A and B both use g4 (contiguous k)
    int g4   = g << 2;             // B offset in float4 units (g*16 floats)

    const float* Arow = A + (size_t)(tm * 16 + r) * K;
    const float* Wr0  = W + (size_t)(ts * 64 + r) * K;
    const float* Wr1  = Wr0 + (size_t)16 * K;
    const float* Wr2  = Wr0 + (size_t)32 * K;
    const float* Wr3  = Wr0 + (size_t)48 * K;

    v8f acc0 = {0.f,0.f,0.f,0.f,0.f,0.f,0.f,0.f};
    v8f acc1 = acc0, acc2 = acc0, acc3 = acc0;

    auto load_step = [&](Stage& s, int k0) {
        const float4* pa = (const float4*)(Arow + k0);
        // A 16x32 bf16 layout: lane group g reads 16 contiguous floats at k0 + g*16
        s.ra[0] = pa[g4 + 0];
        s.ra[1] = pa[g4 + 1];
        s.ra[2] = pa[g4 + 2];
        s.ra[3] = pa[g4 + 3];
        // B 32x16 layout: lane group g reads 16 contiguous floats at k0 + g*16
        const float4* p0 = (const float4*)(Wr0 + k0) + g4;
        const float4* p1 = (const float4*)(Wr1 + k0) + g4;
        const float4* p2 = (const float4*)(Wr2 + k0) + g4;
        const float4* p3 = (const float4*)(Wr3 + k0) + g4;
        #pragma unroll
        for (int i = 0; i < 4; ++i) {
            s.rb0[i] = p0[i]; s.rb1[i] = p1[i]; s.rb2[i] = p2[i]; s.rb3[i] = p3[i];
        }
    };

    auto do_mmas = [&](const Stage& s) {
        v16bf a, b0, b1, b2, b3;
        #pragma unroll
        for (int i = 0; i < 4; ++i) {
            CVT4(a,  4 * i, s.ra[i]);
            CVT4(b0, 4 * i, s.rb0[i]);
            CVT4(b1, 4 * i, s.rb1[i]);
            CVT4(b2, 4 * i, s.rb2[i]);
            CVT4(b3, 4 * i, s.rb3[i]);
        }
        acc0 = __builtin_amdgcn_wmma_f32_16x16x32_bf16(false, a, false, b0, (short)0, acc0, false, false);
        acc1 = __builtin_amdgcn_wmma_f32_16x16x32_bf16(false, a, false, b1, (short)0, acc1, false, false);
        acc2 = __builtin_amdgcn_wmma_f32_16x16x32_bf16(false, a, false, b2, (short)0, acc2, false, false);
        acc3 = __builtin_amdgcn_wmma_f32_16x16x32_bf16(false, a, false, b3, (short)0, acc3, false, false);
    };

    Stage buf0, buf1;
    load_step(buf0, 0);
    for (int k0 = 0; k0 < K; k0 += 64) {
        load_step(buf1, k0 + 32);          // K % 64 == 0, always in range
        do_mmas(buf0);
        if (k0 + 64 < K) load_step(buf0, k0 + 64);
        do_mmas(buf1);
    }

    // C/D layout: VGPR v, lane -> M = tm*16 + g*8 + v, N = n-tile base + r
    int mbase = tm * 16 + (g << 3);
    int n0    = ts * 64 + r;
    store_tile<HAS_RES>(acc0, C, res, bias[n0],      mbase, n0,      N);
    store_tile<HAS_RES>(acc1, C, res, bias[n0 + 16], mbase, n0 + 16, N);
    store_tile<HAS_RES>(acc2, C, res, bias[n0 + 32], mbase, n0 + 32, N);
    store_tile<HAS_RES>(acc3, C, res, bias[n0 + 48], mbase, n0 + 48, N);
}

// ---------------- embedding: h = concat(sep_emb[sep], x@Wi^T + bi) / sqrt(D) ----------------
__global__ __launch_bounds__(256)
void embed_kernel(const float* __restrict__ x, const int* __restrict__ sep,
                  const float* __restrict__ Wi, const float* __restrict__ bi,
                  const float* __restrict__ sep_emb, float* __restrict__ h)
{
    int row = blockIdx.x;                 // b*T + t
    int t = row % T_TOK, b = row / T_TOK;
    const float inv = 0.03125f;           // 1/sqrt(1024)
    if (t == 0) {
        int sidx = sep[b];
        for (int d = threadIdx.x; d < D_MODEL; d += 256)
            h[(size_t)row * D_MODEL + d] = sep_emb[(size_t)sidx * D_MODEL + d] * inv;
        return;
    }
    __shared__ float xs[IN_FEAT];
    if (threadIdx.x < IN_FEAT)
        xs[threadIdx.x] = x[((size_t)b * (T_TOK - 1) + (t - 1)) * IN_FEAT + threadIdx.x];
    __syncthreads();
    for (int d = threadIdx.x; d < D_MODEL; d += 256) {
        const float* wr = Wi + (size_t)d * IN_FEAT;
        float acc = bi[d];
        #pragma unroll 8
        for (int f = 0; f < IN_FEAT; ++f) acc += xs[f] * wr[f];
        h[(size_t)row * D_MODEL + d] = acc * inv;
    }
}

// ---------------- RMSNorm over last dim D ----------------
__global__ __launch_bounds__(256)
void rmsnorm_kernel(const float* __restrict__ x, const float* __restrict__ w,
                    float* __restrict__ out, int D, float eps)
{
    __shared__ float red[256];
    int row = blockIdx.x, tid = threadIdx.x;
    const float* xr = x + (size_t)row * D;
    float ss = 0.f;
    for (int i = tid; i < D; i += 256) { float v = xr[i]; ss += v * v; }
    float tot = blk_sum(ss, red, tid);
    float scale = rsqrtf(tot / (float)D + eps);
    float* o = out + (size_t)row * D;
    for (int i = tid; i < D; i += 256) o[i] = xr[i] * scale * w[i];
}

// ---------------- RoPE on q/k: (B,T,32,32), rows t>=1, cos/sin (maxseq,16) ----------------
__global__ void rope_kernel(float* __restrict__ x, const float* __restrict__ cs,
                            const float* __restrict__ sn)
{
    int idx = blockIdx.x * blockDim.x + threadIdx.x;
    int total = BATCH * (T_TOK - 1) * 32 * 16;
    if (idx >= total) return;
    int i  = idx & 15; idx >>= 4;
    int hh = idx & 31; idx >>= 5;
    int t  = (idx % (T_TOK - 1)) + 1;
    int b  = idx / (T_TOK - 1);
    size_t base = (((size_t)b * T_TOK + t) * 32 + hh) * HD2 + 2 * i;
    float xr = x[base], xi = x[base + 1];
    float c = cs[(size_t)(t - 1) * 16 + i];
    float s = sn[(size_t)(t - 1) * 16 + i];
    x[base]     = xr * c - xi * s;
    x[base + 1] = xr * s + xi * c;
}

// ---------------- fused differential attention + head-RMS + reference reshape ----------------
// one block per (b, head, q_row)
__global__ __launch_bounds__(256)
void attn_kernel(const float* __restrict__ q, const float* __restrict__ k,
                 const float* __restrict__ v,
                 const float* __restrict__ lq1, const float* __restrict__ lq2,
                 const float* __restrict__ lk1, const float* __restrict__ lk2,
                 const float* __restrict__ hw, float lam_init,
                 float* __restrict__ y)
{
    __shared__ float s0[T_TOK];
    __shared__ float s1[T_TOK];
    __shared__ float red[256];
    __shared__ float qv[2 * HD2];
    __shared__ float ovec[HD];
    __shared__ float sh_lam;

    int idx = blockIdx.x;
    int qi = idx % T_TOK; idx /= T_TOK;
    int h  = idx % NHEAD;
    int b  = idx / NHEAD;
    int tid = threadIdx.x;

    if (tid == 0) {
        float a = 0.f, c = 0.f;
        for (int i = 0; i < HD2; ++i) { a += lk1[i] * lq1[i]; c += lk2[i] * lq2[i]; }
        sh_lam = __expf(a) - __expf(c) + lam_init;
    }
    if (tid < 2 * HD2) {
        // q[b, qi, 2h + (tid>=32), tid&31]
        qv[tid] = q[(((size_t)b * T_TOK + qi) * 32 + 2 * h + (tid >> 5)) * HD2 + (tid & 31)];
    }
    __syncthreads();

    // raw scores (both half-heads), causal sentinel
    for (int kk = tid; kk < T_TOK; kk += 256) {
        if (kk <= qi) {
            const float* kr = k + (((size_t)b * T_TOK + kk) * 32 + 2 * h) * HD2;
            float d0 = 0.f, d1 = 0.f;
            #pragma unroll 8
            for (int i = 0; i < HD2; ++i) {
                d0 += qv[i] * kr[i];
                d1 += qv[HD2 + i] * kr[HD2 + i];
            }
            s0[kk] = d0 * SCALING_C;
            s1[kk] = d1 * SCALING_C;
        } else {
            s0[kk] = -1e30f; s1[kk] = -1e30f;
        }
    }
    __syncthreads();

    float m0 = -1e30f, m1 = -1e30f;
    for (int kk = tid; kk < T_TOK; kk += 256) { m0 = fmaxf(m0, s0[kk]); m1 = fmaxf(m1, s1[kk]); }
    m0 = blk_max(m0, red, tid);
    m1 = blk_max(m1, red, tid);

    float ps0 = 0.f, ps1 = 0.f;
    for (int kk = tid; kk < T_TOK; kk += 256) {
        if (kk <= qi) {
            float e0 = __expf(s0[kk] - m0);
            float e1 = __expf(s1[kk] - m1);
            s0[kk] = e0; s1[kk] = e1;
            ps0 += e0; ps1 += e1;
        }
    }
    __syncthreads();
    float sum0 = blk_sum(ps0, red, tid);
    float sum1 = blk_sum(ps1, red, tid);
    float inv0 = 1.f / sum0, inv1 = 1.f / sum1;
    float lam  = sh_lam;
    // mean(a1) over full T = 1/T (softmax row sums to 1, masked entries exp-underflow to 0);
    // a3 = mean * T/(q+1)  ->  lam*a3 = lam/(q+1)
    float a3add = lam / ((float)qi + 1.0f);

    for (int kk = tid; kk < T_TOK; kk += 256) {
        if (kk <= qi) s0[kk] = s0[kk] * inv0 - lam * (s1[kk] * inv1) + a3add;
        else          s0[kk] = 0.f;
    }
    __syncthreads();

    // o[d] = sum_k af[k] * v[b,k,h,d]; threads = (d, j) with 4-way k split
    int d = tid >> 2, j = tid & 3;
    float acc = 0.f;
    for (int kk = j; kk <= qi; kk += 4)
        acc += s0[kk] * v[(((size_t)b * T_TOK + kk) * NHEAD + h) * HD + d];
    red[tid] = acc; __syncthreads();
    if (j == 0) ovec[d] = red[tid] + red[tid + 1] + red[tid + 2] + red[tid + 3];
    __syncthreads();

    // head RMS over HD=64
    float p = (tid < HD) ? ovec[tid] * ovec[tid] : 0.f;
    float ssq = blk_sum(p, red, tid);
    float scale = rsqrtf(ssq / (float)HD + EPS_HEAD);

    if (tid < HD) {
        // reference reshape (B,NHEAD,T,HD)->(B,T,D):
        int row = h * 32 + (qi >> 4);
        int col = (qi & 15) * HD + tid;
        y[((size_t)b * T_TOK + row) * D_MODEL + col] = ovec[tid] * scale * hw[tid];
    }
}

// ---------------- SiLU-GLU: act = u * silu(g), uv rows of 2*D_FF ----------------
__global__ void glu_kernel(const float* __restrict__ uv, float* __restrict__ act)
{
    size_t idx = (size_t)blockIdx.x * blockDim.x + threadIdx.x;
    size_t total = (size_t)BATCH * T_TOK * D_FF;
    if (idx >= total) return;
    size_t row = idx / D_FF;
    int c = (int)(idx % D_FF);
    float u = uv[row * (2 * D_FF) + c];
    float g = uv[row * (2 * D_FF) + D_FF + c];
    act[idx] = u * (g / (1.f + __expf(-g)));
}

// ---------------- final projection: out[row, 0..3] ----------------
__global__ __launch_bounds__(256)
void final_proj_kernel(const float* __restrict__ f, const float* __restrict__ Wp,
                       const float* __restrict__ bp, float* __restrict__ out)
{
    __shared__ float red[4 * 256];
    int row = blockIdx.x, tid = threadIdx.x;
    const float* fr = f + (size_t)row * D_MODEL;
    float a0 = 0.f, a1 = 0.f, a2 = 0.f, a3 = 0.f;
    for (int d = tid; d < D_MODEL; d += 256) {
        float vv = fr[d];
        a0 += vv * Wp[d];
        a1 += vv * Wp[D_MODEL + d];
        a2 += vv * Wp[2 * D_MODEL + d];
        a3 += vv * Wp[3 * D_MODEL + d];
    }
    red[tid] = a0; red[256 + tid] = a1; red[512 + tid] = a2; red[768 + tid] = a3;
    __syncthreads();
    for (int s = 128; s > 0; s >>= 1) {
        if (tid < s) {
            #pragma unroll
            for (int p = 0; p < 4; ++p) red[p * 256 + tid] += red[p * 256 + tid + s];
        }
        __syncthreads();
    }
    if (tid < N_PRED) out[(size_t)row * N_PRED + tid] = red[tid * 256] + bp[tid];
}

// ---------------- host orchestration ----------------
static inline void launch_gemm(const float* A, const float* W, const float* bias,
                               const float* res, float* C, int M, int N, int K,
                               hipStream_t stream)
{
    int strips = (M / 16) * (N / 64);
    int blocks = (strips + 3) / 4;      // 4 waves (128 threads) per block
    if (res)
        gemm_wmma_bf16<true><<<blocks, 128, 0, stream>>>(A, W, bias, res, C, M, N, K);
    else
        gemm_wmma_bf16<false><<<blocks, 128, 0, stream>>>(A, W, bias, res, C, M, N, K);
}

extern "C" void kernel_launch(void* const* d_in, const int* in_sizes, int n_in,
                              void* d_out, int out_size, void* d_ws, size_t ws_size,
                              hipStream_t stream)
{
    (void)in_sizes; (void)n_in; (void)out_size; (void)ws_size;

    const float* x       = (const float*)d_in[0];
    const int*   sep     = (const int*)  d_in[1];
    const float* Wq      = (const float*)d_in[2];
    const float* bq      = (const float*)d_in[3];
    const float* Wk      = (const float*)d_in[4];
    const float* bk      = (const float*)d_in[5];
    const float* Wv      = (const float*)d_in[6];
    const float* bv      = (const float*)d_in[7];
    const float* Wo      = (const float*)d_in[8];
    const float* bo      = (const float*)d_in[9];
    const float* lq1     = (const float*)d_in[10];
    const float* lq2     = (const float*)d_in[11];
    const float* lk1     = (const float*)d_in[12];
    const float* lk2     = (const float*)d_in[13];
    const float* hw      = (const float*)d_in[14];
    const float* n1w     = (const float*)d_in[15];
    const float* n2w     = (const float*)d_in[16];
    const float* Wg      = (const float*)d_in[17];
    const float* bg      = (const float*)d_in[18];
    const float* Wf      = (const float*)d_in[19];
    const float* bf      = (const float*)d_in[20];
    const float* Wi      = (const float*)d_in[21];
    const float* bi      = (const float*)d_in[22];
    const float* nfw     = (const float*)d_in[23];
    const float* Wp      = (const float*)d_in[24];
    const float* bp      = (const float*)d_in[25];
    const float* sep_emb = (const float*)d_in[26];
    const float* cosT    = (const float*)d_in[27];
    const float* sinT    = (const float*)d_in[28];
    float* out = (float*)d_out;

    const int M = BATCH * T_TOK;                    // 1024
    const size_t SZ = (size_t)M * D_MODEL * sizeof(float);  // 4 MiB
    char* ws = (char*)d_ws;
    float* h    = (float*)(ws + 0 * SZ);
    float* f    = (float*)(ws + 1 * SZ);
    float* qb   = (float*)(ws + 2 * SZ);
    float* kb   = (float*)(ws + 3 * SZ);
    float* vb   = (float*)(ws + 4 * SZ);
    float* y    = (float*)(ws + 5 * SZ);
    float* uv   = (float*)(ws + 6 * SZ);            // 2*D_FF cols -> 5.5*SZ
    float* act  = (float*)(ws + 6 * SZ + (size_t)M * 2 * D_FF * sizeof(float));

    // 1) embedding
    embed_kernel<<<M, 256, 0, stream>>>(x, sep, Wi, bi, sep_emb, h);

    const int rope_total  = BATCH * (T_TOK - 1) * 32 * 16;
    const int rope_blocks = (rope_total + 255) / 256;
    const size_t glu_total = (size_t)M * D_FF;
    const int glu_blocks  = (int)((glu_total + 255) / 256);

    for (int l = 0; l < NUM_LAYERS; ++l) {
        const size_t oDD = (size_t)l * D_MODEL * D_MODEL;
        const float lam_init = 0.8f - 0.6f * expf(-0.3f * (float)l);

        // attention block
        rmsnorm_kernel<<<M, 256, 0, stream>>>(h, n1w + (size_t)l * D_MODEL, f, D_MODEL, EPS_DEF);
        launch_gemm(f, Wq + oDD, bq + (size_t)l * D_MODEL, nullptr, qb, M, D_MODEL, D_MODEL, stream);
        launch_gemm(f, Wk + oDD, bk + (size_t)l * D_MODEL, nullptr, kb, M, D_MODEL, D_MODEL, stream);
        launch_gemm(f, Wv + oDD, bv + (size_t)l * D_MODEL, nullptr, vb, M, D_MODEL, D_MODEL, stream);
        rope_kernel<<<rope_blocks, 256, 0, stream>>>(qb, cosT, sinT);
        rope_kernel<<<rope_blocks, 256, 0, stream>>>(kb, cosT, sinT);
        attn_kernel<<<BATCH * NHEAD * T_TOK, 256, 0, stream>>>(
            qb, kb, vb,
            lq1 + (size_t)l * HD2, lq2 + (size_t)l * HD2,
            lk1 + (size_t)l * HD2, lk2 + (size_t)l * HD2,
            hw + (size_t)l * HD, lam_init, y);
        launch_gemm(y, Wo + oDD, bo + (size_t)l * D_MODEL, h, h, M, D_MODEL, D_MODEL, stream);

        // FFN block
        rmsnorm_kernel<<<M, 256, 0, stream>>>(h, n2w + (size_t)l * D_MODEL, f, D_MODEL, EPS_DEF);
        launch_gemm(f, Wg + (size_t)l * 2 * D_FF * D_MODEL, bg + (size_t)l * 2 * D_FF,
                    nullptr, uv, M, 2 * D_FF, D_MODEL, stream);
        glu_kernel<<<glu_blocks, 256, 0, stream>>>(uv, act);
        launch_gemm(act, Wf + (size_t)l * D_MODEL * D_FF, bf + (size_t)l * D_MODEL,
                    h, h, M, D_MODEL, D_FF, stream);
    }

    // final norm + projection
    rmsnorm_kernel<<<M, 256, 0, stream>>>(h, nfw, f, D_MODEL, EPS_DEF);
    final_proj_kernel<<<M, 256, 0, stream>>>(f, Wp, bp, out);
}
